// AtomicChainModel_3599182594913
// MI455X (gfx1250) — compile-verified, matrix operand order checked by reading, not run on hardware
//
#include <hip/hip_runtime.h>
#include <hip/hip_bf16.h>

#define B_ 32
#define N_ 128
#define D_ 64
#define H_ 128
#define LL_ 3
#define OUT_ 1300

typedef __attribute__((ext_vector_type(16))) _Float16 v16h;
typedef __attribute__((ext_vector_type(8)))  float    v8f;

// K index inside a 16x32 f16 WMMA operand for half-slot t (0..15) of a lane.
// From ISA 7.12.2: VGPR v=t/2 holds K pairs; lanes>=16 shifted by +8; v>=4 adds +16.
__device__ __forceinline__ int wmma_k_of(int t, int lane) {
  int v = t >> 1, p = t & 1;
  return ((v & 4) ? 16 : 0) + ((lane & 16) ? 8 : 0) + ((v & 3) << 1) + p;
}

union HUpk { _Float16 h[2]; unsigned u; };

// ---------------- prep: node0 = emb[type] * mask ----------------
__global__ void prep_kernel(const float* __restrict__ atoms,
                            const float* __restrict__ emb,
                            float* __restrict__ nodeA) {
  int gid = blockIdx.x * 256 + threadIdx.x;           // over B*N*D
  if (gid >= B_ * N_ * D_) return;
  int bn = gid / D_, d = gid % D_;
  float ty = atoms[(size_t)bn * 2 + 0];
  float m = (ty >= 0.f) ? 1.f : 0.f;
  int t = (int)ty; t = t < 0 ? 0 : (t > 1 ? 1 : t);
  nodeA[gid] = emb[t * D_ + d] * m;
}

// ------- wprep: M_l = dw2 @ We_l and Wr_l, both stored in WMMA B-fragment
//         layout (f16); bias_l = mb1_l + db2 @ We_l -------
// Fragment layout: idx = ((l*8 + h/16)*2 + d/32)*32 + lane, 16 halves each.
__global__ __launch_bounds__(128) void wprep_kernel(
    const float* __restrict__ dw2, const float* __restrict__ db2,
    const float* __restrict__ mw1, const float* __restrict__ mb1,
    _Float16* __restrict__ Mfrag, _Float16* __restrict__ Wrfrag,
    float* __restrict__ biasArr) {
  int l = blockIdx.x, h = threadIdx.x;
  const float* Wr = mw1 + (size_t)l * 192 * H_ + D_ * H_;      // rows 64..127
  const float* We = mw1 + (size_t)l * 192 * H_ + 2 * D_ * H_;  // rows 128..191
  float bias = mb1[l * H_ + h];
  for (int e = 0; e < D_; ++e) bias = fmaf(db2[e], We[e * H_ + h], bias);
  biasArr[l * H_ + h] = bias;
  for (int d = 0; d < D_; ++d) {
    float m = 0.f;
    for (int e = 0; e < D_; ++e) m = fmaf(dw2[d * D_ + e], We[e * H_ + h], m);
    int c = d >> 5, dd = d & 31;
    int t    = ((dd & 16) ? 8 : 0) + (dd & 7);
    int lane = (h & 15) + ((dd & 8) ? 16 : 0);
    size_t idx = (size_t)(((l * 8 + (h >> 4)) * 2 + c) * 32 + lane) * 16 + t;
    Mfrag[idx]  = (_Float16)m;
    Wrfrag[idx] = (_Float16)Wr[d * H_ + h];
  }
}

// ------- nf: swizzle node[b] into f16 WMMA A-fragment layout (16 slabs x 32 x 16) ----
__global__ __launch_bounds__(256) void nf_kernel(
    const float* __restrict__ node_in, _Float16* __restrict__ nodeFrag) {
  int b = blockIdx.x, tid = threadIdx.x;
  for (int s = tid; s < 512; s += 256) {
    int slab = s >> 5, lane2 = s & 31;
    int jt = slab >> 1, cch = slab & 1;
    int j = jt * 16 + (lane2 & 15);
    const float* row = node_in + ((size_t)b * N_ + j) * D_ + cch * 32;
    unsigned w[8];
#pragma unroll
    for (int q = 0; q < 8; ++q) {
      int k0 = wmma_k_of(2 * q, lane2);   // k1 = k0 + 1
      HUpk pk;
      pk.h[0] = (_Float16)row[k0];
      pk.h[1] = (_Float16)row[k0 + 1];
      w[q] = pk.u;
    }
    uint4* dst = (uint4*)(nodeFrag + ((size_t)b * 512 + s) * 16);
    dst[0] = make_uint4(w[0], w[1], w[2], w[3]);
    dst[1] = make_uint4(w[4], w[5], w[6], w[7]);
  }
}

// ------------- main fused layer: K=128 WMMA ([g | node] @ [M ; Wr]), folded msg/agg ----
__global__ __launch_bounds__(256) void layer_kernel(
    const float* __restrict__ node_in, float* __restrict__ node_out,
    const _Float16* __restrict__ nodeFrag, const float* __restrict__ atoms,
    const _Float16* __restrict__ Mfrag, const _Float16* __restrict__ Wrfrag,
    const float* __restrict__ biasArr,
    const float* __restrict__ mw1, const float* __restrict__ mw2,
    const float* __restrict__ mb2, const float* __restrict__ uw,
    const float* __restrict__ ub, const float* __restrict__ dw1,
    const float* __restrict__ db1, const float* __restrict__ ucl, int layer) {
  extern __shared__ char smem[];
  _Float16* sG    = (_Float16*)smem;                     // g A-frags: 16KB
  _Float16* sNF   = (_Float16*)(smem + 16384);           // node A-frags: 16KB
  float*    sS    = (float*)(smem + 32768);              // 128
  float*    sPos  = sS + 128;                            // 128
  float*    sWj   = sPos + 128;                          // 128  (mask_j * (j != i))
  float*    sDw1  = sWj + 128;                           // 64
  float*    sDb1  = sDw1 + 64;                           // 64
  float*    sNode = sDb1 + 64;                           // 64
  float*    sHsum = sNode + 64;                          // 128
  float*    sAgg  = sHsum + 128;                         // 64

  const int i = blockIdx.x, b = blockIdx.y, tid = threadIdx.x;
  const float Lb = ucl[b];

  // ---- async-stage nodeFrag[b] (16KB) straight into LDS (ASYNCcnt path) ----
  {
    unsigned ldsN = (unsigned)(size_t)sNF + (unsigned)tid * 16u;
    uint64_t gN = (uint64_t)(size_t)(nodeFrag + (size_t)b * 512 * 16) + (uint64_t)tid * 16u;
#pragma unroll
    for (int it = 0; it < 4; ++it) {
      asm volatile("global_load_async_to_lds_b128 %0, %1, off"
                   :: "v"(ldsN + (unsigned)it * 4096u), "v"(gN + (uint64_t)it * 4096u)
                   : "memory");
    }
  }
  if (tid < N_) {
    float ty = atoms[((size_t)b * N_ + tid) * 2 + 0];
    sPos[tid] = atoms[((size_t)b * N_ + tid) * 2 + 1];
    sWj[tid]  = ((ty >= 0.f) ? 1.f : 0.f) * ((tid == i) ? 0.f : 1.f);
  }
  if (tid < D_) {
    sDw1[tid]  = dw1[tid];
    sDb1[tid]  = db1[tid];
    sNode[tid] = node_in[((size_t)b * N_ + i) * D_ + tid];
  }
  __syncthreads();  // small arrays visible; async loads still in flight

  if (tid < H_) {  // S[i,h] = node_i @ Ws
    const float* Ws = mw1 + (size_t)layer * 192 * H_;  // rows 0..63
    float a = 0.f;
    for (int d = 0; d < D_; ++d) a = fmaf(sNode[d], Ws[d * H_ + tid], a);
    sS[tid] = a;
  }
  // ---- build g once per workgroup, pre-swizzled in A-fragment layout ----
  {
    const float pos_i = sPos[i];
    for (int s = tid; s < 512; s += 256) {
      int slab = s >> 5, lane2 = s & 31;
      int jt = slab >> 1, cch = slab & 1;
      int j = jt * 16 + (lane2 & 15);
      float dd = fabsf(pos_i - sPos[j]);
      float x  = fminf(dd, Lb - dd) / Lb;
      unsigned w[8];
#pragma unroll
      for (int q = 0; q < 8; ++q) {
        int k0 = cch * 32 + wmma_k_of(2 * q, lane2);   // k1 = k0 + 1
        HUpk pk;
        pk.h[0] = (_Float16)fmaxf(fmaf(x, sDw1[k0],     sDb1[k0]),     0.f);
        pk.h[1] = (_Float16)fmaxf(fmaf(x, sDw1[k0 + 1], sDb1[k0 + 1]), 0.f);
        w[q] = pk.u;
      }
      uint4* dst = (uint4*)(sG + (size_t)s * 16);
      dst[0] = make_uint4(w[0], w[1], w[2], w[3]);
      dst[1] = make_uint4(w[4], w[5], w[6], w[7]);
    }
  }
  asm volatile("s_wait_asynccnt 0" ::: "memory");  // own async LDS writes done
  __syncthreads();                                 // everyone's sNF/sG/sS visible

  // ---- phase 1: WMMA (K=128 = g(64) | node(64)), all 256 threads uniform ----
  const int wave = tid >> 5, lane = tid & 31, ln = lane & 15;
  const int hcol = wave * 16 + ln;

  // B fragments: direct 32B global loads from pre-swizzled weights (L2-hot)
  const v16h* Mv = (const v16h*)Mfrag;
  const v16h* Wv = (const v16h*)Wrfrag;
  const v16h bf0 = Mv[(size_t)(((layer * 8 + wave) * 2 + 0) * 32 + lane)];
  const v16h bf1 = Mv[(size_t)(((layer * 8 + wave) * 2 + 1) * 32 + lane)];
  const v16h bf2 = Wv[(size_t)(((layer * 8 + wave) * 2 + 0) * 32 + lane)];
  const v16h bf3 = Wv[(size_t)(((layer * 8 + wave) * 2 + 1) * 32 + lane)];

  const float sbias = sS[hcol] + biasArr[layer * H_ + hcol];
  const int   rofs  = (lane & 16) ? 8 : 0;
  const v16h*  sGv  = (const v16h*)sG;
  const v16h*  sNv  = (const v16h*)sNF;
  const float* wbase = sWj + rofs;              // + (jt*16+r) as immediates
  float acc = 0.f;

#pragma unroll
  for (int jt = 0; jt < 8; ++jt) {
    v16h ag0 = sGv[(jt * 2 + 0) * 32 + lane];   // 2x ds_load_b128 each
    v16h ag1 = sGv[(jt * 2 + 1) * 32 + lane];
    v16h an0 = sNv[(jt * 2 + 0) * 32 + lane];
    v16h an1 = sNv[(jt * 2 + 1) * 32 + lane];
    v8f c = {};
    c = __builtin_amdgcn_wmma_f32_16x16x32_f16(false, ag0, false, bf0, (short)0, c, false, false);
    c = __builtin_amdgcn_wmma_f32_16x16x32_f16(false, ag1, false, bf1, (short)0, c, false, false);
    c = __builtin_amdgcn_wmma_f32_16x16x32_f16(false, an0, false, bf2, (short)0, c, false, false);
    c = __builtin_amdgcn_wmma_f32_16x16x32_f16(false, an1, false, bf3, (short)0, c, false, false);
#pragma unroll
    for (int r = 0; r < 8; ++r) {  // row-sum over j with adj weights
      float hv = fmaxf(c[r] + sbias, 0.f);
      acc = fmaf(hv, wbase[jt * 16 + r], acc);
    }
  }
  acc += __shfl_xor(acc, 16, 32);   // lanes l and l+16 share the same h column
  if (lane < 16) sHsum[hcol] = acc;
  __syncthreads();

  // ---- phase 2: agg = (hsum @ mw2 + deg*mb2) * m_i ----
  const float tyi = atoms[((size_t)b * N_ + i) * 2 + 0];
  const float m_i = (tyi >= 0.f) ? 1.f : 0.f;
  if (tid < D_) {
    float deg = 0.f;
    for (int j = 0; j < N_; ++j) deg += sWj[j];
    const float* W2 = mw2 + (size_t)layer * H_ * D_;
    float a = 0.f;
    for (int h = 0; h < H_; ++h) a = fmaf(sHsum[h], W2[h * D_ + tid], a);
    sAgg[tid] = (a + deg * mb2[layer * D_ + tid]) * m_i;
  }
  __syncthreads();

  // ---- phase 3: node update ----
  if (tid < D_) {
    const float* U = uw + (size_t)layer * 2 * D_ * D_;
    float u = ub[layer * D_ + tid];
    for (int d2 = 0; d2 < D_; ++d2) u = fmaf(sNode[d2], U[d2 * D_ + tid], u);
    for (int d2 = 0; d2 < D_; ++d2) u = fmaf(sAgg[d2], U[(D_ + d2) * D_ + tid], u);
    node_out[((size_t)b * N_ + i) * D_ + tid] = fmaxf(u, 0.f) * m_i;
  }
}

// ------------- head MLP (tiny, VALU) -------------
__global__ __launch_bounds__(256) void head_kernel(
    const float* __restrict__ node, const float* __restrict__ ucl,
    const float* __restrict__ ow1, const float* __restrict__ ob1,
    const float* __restrict__ ow2, const float* __restrict__ ob2,
    const float* __restrict__ ow3, const float* __restrict__ ob3,
    float* __restrict__ out) {
  __shared__ float sC[65];
  __shared__ float sO1[128];
  __shared__ float sO2[256];
  int b = blockIdx.x, tid = threadIdx.x;
  if (tid < D_) {
    float s = 0.f;
    for (int i2 = 0; i2 < N_; ++i2) s += node[((size_t)b * N_ + i2) * D_ + tid];
    sC[tid] = s;
  }
  if (tid == 64) sC[64] = ucl[b];
  __syncthreads();
  if (tid < 128) {
    float a = ob1[tid];
    for (int k = 0; k < 65; ++k) a = fmaf(sC[k], ow1[k * 128 + tid], a);
    sO1[tid] = fmaxf(a, 0.f);
  }
  __syncthreads();
  {
    float a = ob2[tid];
    for (int k = 0; k < 128; ++k) a = fmaf(sO1[k], ow2[k * 256 + tid], a);
    sO2[tid] = fmaxf(a, 0.f);
  }
  __syncthreads();
  for (int o = tid; o < OUT_; o += 256) {
    float a = ob3[o];
    for (int k = 0; k < 256; ++k) a = fmaf(sO2[k], ow3[k * OUT_ + o], a);
    out[(size_t)b * OUT_ + o] = a;
  }
}

extern "C" void kernel_launch(void* const* d_in, const int* in_sizes, int n_in,
                              void* d_out, int out_size, void* d_ws, size_t ws_size,
                              hipStream_t stream) {
  const float* ucl   = (const float*)d_in[0];
  const float* atoms = (const float*)d_in[1];
  const float* emb   = (const float*)d_in[2];
  const float* dw1   = (const float*)d_in[3];
  const float* db1   = (const float*)d_in[4];
  const float* dw2   = (const float*)d_in[5];
  const float* db2   = (const float*)d_in[6];
  const float* mw1   = (const float*)d_in[7];
  const float* mb1   = (const float*)d_in[8];
  const float* mw2   = (const float*)d_in[9];
  const float* mb2   = (const float*)d_in[10];
  const float* uw    = (const float*)d_in[11];
  const float* ub    = (const float*)d_in[12];
  const float* ow1   = (const float*)d_in[13];
  const float* ob1   = (const float*)d_in[14];
  const float* ow2   = (const float*)d_in[15];
  const float* ob2   = (const float*)d_in[16];
  const float* ow3   = (const float*)d_in[17];
  const float* ob3   = (const float*)d_in[18];
  float* out = (float*)d_out;

  char* ws = (char*)d_ws;
  float*    nodeA    = (float*)(ws);                                // 1 MB
  float*    nodeB    = (float*)(ws + (1u << 20));                   // 1 MB
  _Float16* nodeFrag = (_Float16*)(ws + (2u << 20));                // 512 KB, A-frag layout
  _Float16* Mfrag    = (_Float16*)(ws + (2u << 20) + 524288);       // 48 KB, B-frag layout
  _Float16* Wrfrag   = (_Float16*)(ws + (2u << 20) + 524288 + 49152);  // 48 KB
  float*    biasArr  = (float*)(ws + (2u << 20) + 524288 + 98304);  // 1.5 KB

  prep_kernel<<<(B_ * N_ * D_ + 255) / 256, 256, 0, stream>>>(atoms, emb, nodeA);
  wprep_kernel<<<LL_, 128, 0, stream>>>(dw2, db2, mw1, mb1, Mfrag, Wrfrag, biasArr);

  const int SH = 32768 + (128 + 128 + 128 + 64 + 64 + 64 + 128 + 64) * 4;
  float* cur = nodeA; float* nxt = nodeB;
  for (int l = 0; l < LL_; ++l) {
    nf_kernel<<<B_, 256, 0, stream>>>(cur, nodeFrag);
    layer_kernel<<<dim3(N_, B_), 256, SH, stream>>>(cur, nxt, nodeFrag, atoms, Mfrag,
                                                    Wrfrag, biasArr, mw1, mw2, mb2,
                                                    uw, ub, dw1, db1, ucl, l);
    float* t = cur; cur = nxt; nxt = t;
  }
  head_kernel<<<B_, 256, 0, stream>>>(cur, ucl, ow1, ob1, ow2, ob2, ow3, ob3, out);
}
